// GATModel_85521388798597
// MI455X (gfx1250) — matrix operand order repaired and checked by reading, not compiled, not run on hardware
//
#include <hip/hip_runtime.h>
#include <hip/hip_fp16.h>

typedef __attribute__((ext_vector_type(16))) _Float16 v16h;
typedef __attribute__((ext_vector_type(8)))  float    v8f;

#define NN      50000           // nodes
#define NE      800000          // edges (before self loops)
#define ETOT    (NE + NN)       // edges incl. self loops
#define FDIM    128             // every layer is 128 -> 128
#define NEG_SLOPE 0.2f
#define KEY_NEG 0x00800000u     // fkey(-FLT_MAX)

// ---- monotonic uint key for float atomicMax ----
__device__ __forceinline__ unsigned fkey(float f) {
    unsigned u = __float_as_uint(f);
    return (u & 0x80000000u) ? ~u : (u | 0x80000000u);
}
__device__ __forceinline__ float funkey(unsigned k) {
    return (k & 0x80000000u) ? __uint_as_float(k & 0x7fffffffu)
                             : __uint_as_float(~k);
}

__device__ __forceinline__ void edge_ends(int e, const int* __restrict__ ei,
                                          int& s, int& d) {
    if (e < NE) { s = ei[e]; d = ei[NE + e]; }
    else        { s = e - NE; d = e - NE; }          // self loop
}

// ============================================================
// Pack W (fp32, [K=128][N=128] row-major) into the WMMA B-operand
// lane layout for v_wmma_f32_16x16x32_f16:
//   lane = khalf*16 + n (n=0..15), element j <-> K = kstep*32 + khalf*8 + (j<8 ? j : j+8)
// Wp layout: [ntile(8)][kstep(4)][lane(32)][16 halfs]  (32 KB)
// ============================================================
__global__ void pack_w_kernel(const float* __restrict__ W,
                              _Float16* __restrict__ Wp) {
    int t = blockIdx.x * blockDim.x + threadIdx.x;      // 0..1023
    if (t >= 8 * 4 * 32) return;
    int lane  = t & 31;
    int kstep = (t >> 5) & 3;
    int ntile = t >> 7;
    int nl = lane & 15, khalf = lane >> 4;
    int col   = ntile * 16 + nl;
    int kbase = kstep * 32 + khalf * 8;
    _Float16* dst = Wp + t * 16;
#pragma unroll
    for (int j = 0; j < 16; ++j)
        dst[j] = (_Float16)W[(kbase + (j < 8 ? j : j + 8)) * FDIM + col];
}

// ============================================================
// GEMM  Y[50000x128] = X[50000x128] @ W  via v_wmma_f32_16x16x32_f16
// block = 256 threads = 8 waves; wave w owns N-tile w; grid = 3125 M-tiles.
// A layout per ISA: lanes 0-15 -> rows M=0..15, elements 0-7 = K kb..kb+7,
// elements 8-15 = K kb+16..kb+23, kb = kstep*32 + khalf*8.
// ============================================================
__global__ void gemm128_kernel(const float* __restrict__ X,
                               const _Float16* __restrict__ Wp,
                               float* __restrict__ Y) {
    int wave = threadIdx.x >> 5;            // n-tile 0..7
    int lane = threadIdx.x & 31;
    int nl = lane & 15, khalf = lane >> 4;
    int row = blockIdx.x * 16 + nl;
    const float* xr = X + (size_t)row * FDIM;

    v8f c = {};
#pragma unroll
    for (int kstep = 0; kstep < 4; ++kstep) {
        int b0 = kstep * 32 + khalf * 8;
        float4 f0 = *(const float4*)(xr + b0);
        float4 f1 = *(const float4*)(xr + b0 + 4);
        float4 f2 = *(const float4*)(xr + b0 + 16);
        float4 f3 = *(const float4*)(xr + b0 + 20);
        v16h a;
        a[0]=(_Float16)f0.x; a[1]=(_Float16)f0.y; a[2]=(_Float16)f0.z; a[3]=(_Float16)f0.w;
        a[4]=(_Float16)f1.x; a[5]=(_Float16)f1.y; a[6]=(_Float16)f1.z; a[7]=(_Float16)f1.w;
        a[8]=(_Float16)f2.x; a[9]=(_Float16)f2.y; a[10]=(_Float16)f2.z; a[11]=(_Float16)f2.w;
        a[12]=(_Float16)f3.x; a[13]=(_Float16)f3.y; a[14]=(_Float16)f3.z; a[15]=(_Float16)f3.w;
        v16h b = *(const v16h*)(Wp + (((wave * 4 + kstep) * 32 + lane) << 4));
        c = __builtin_amdgcn_wmma_f32_16x16x32_f16(false, a, false, b,
                                                   (short)0, c, false, false);
    }
    // D layout: VGPR r -> row r (lanes 0-15) / row r+8 (lanes 16-31), col = lane%16
    int col   = wave * 16 + nl;
    int rbase = blockIdx.x * 16 + khalf * 8;
#pragma unroll
    for (int r = 0; r < 8; ++r)
        Y[(size_t)(rbase + r) * FDIM + col] = c[r];
}

// ============================================================
// Per-node attention scores  a_s[n,h] = <h[n,h,:], att_src[h,:]>
// ============================================================
template <int HEADS, int C>
__global__ void att_scores_kernel(const float* __restrict__ H,
                                  const float* __restrict__ asrc,
                                  const float* __restrict__ adst,
                                  float* __restrict__ a_s,
                                  float* __restrict__ a_d) {
    int t = blockIdx.x * blockDim.x + threadIdx.x;
    if (t >= NN * HEADS) return;
    int n = t / HEADS, h = t % HEADS;
    const float* hp = H + (size_t)n * FDIM + h * C;
    float ss = 0.f, sd = 0.f;
#pragma unroll 4
    for (int c = 0; c < C; ++c) {
        float v = hp[c];
        ss += v * asrc[h * C + c];
        sd += v * adst[h * C + c];
    }
    a_s[t] = ss; a_d[t] = sd;
}

// zero accumulators for a layer (harness poisons ws -> must re-init every call)
template <int HEADS>
__global__ void init_layer_kernel(unsigned* __restrict__ m,
                                  float* __restrict__ denom,
                                  float* __restrict__ outbuf) {
    int t = blockIdx.x * blockDim.x + threadIdx.x;
    if (t < NN * FDIM) outbuf[t] = 0.f;
    if (t < NN * HEADS) { m[t] = KEY_NEG; denom[t] = 0.f; }
}

template <int HEADS>
__global__ void edge_max_kernel(const int* __restrict__ ei,
                                const float* __restrict__ a_s,
                                const float* __restrict__ a_d,
                                unsigned* __restrict__ m) {
    int e = blockIdx.x * blockDim.x + threadIdx.x;
    if (e >= ETOT) return;
    int s, d; edge_ends(e, ei, s, d);
#pragma unroll
    for (int h = 0; h < HEADS; ++h) {
        float v = a_s[s * HEADS + h] + a_d[d * HEADS + h];
        v = v > 0.f ? v : NEG_SLOPE * v;
        atomicMax(&m[d * HEADS + h], fkey(v));
    }
}

template <int HEADS>
__global__ void edge_denom_kernel(const int* __restrict__ ei,
                                  const float* __restrict__ a_s,
                                  const float* __restrict__ a_d,
                                  const unsigned* __restrict__ m,
                                  float* __restrict__ denom) {
    int e = blockIdx.x * blockDim.x + threadIdx.x;
    if (e >= ETOT) return;
    int s, d; edge_ends(e, ei, s, d);
#pragma unroll
    for (int h = 0; h < HEADS; ++h) {
        float v = a_s[s * HEADS + h] + a_d[d * HEADS + h];
        v = v > 0.f ? v : NEG_SLOPE * v;
        atomicAdd(&denom[d * HEADS + h], expf(v - funkey(m[d * HEADS + h])));
    }
}

// 4 threads per edge, 32 channels each: out[dst] += alpha * H[src]
template <int HEADS, int C>
__global__ void edge_scatter_kernel(const int* __restrict__ ei,
                                    const float* __restrict__ a_s,
                                    const float* __restrict__ a_d,
                                    const unsigned* __restrict__ m,
                                    const float* __restrict__ denom,
                                    const float* __restrict__ H,
                                    float* __restrict__ out) {
    int t = blockIdx.x * blockDim.x + threadIdx.x;
    if (t >= ETOT * 4) return;
    int e = t >> 2, q = t & 3;
    int s, d; edge_ends(e, ei, s, d);
    int c0 = q * 32;
    int h0 = c0 / C, h1 = (c0 + 31) / C;
    for (int h = h0; h <= h1; ++h) {
        float v = a_s[s * HEADS + h] + a_d[d * HEADS + h];
        v = v > 0.f ? v : NEG_SLOPE * v;
        float alpha = expf(v - funkey(m[d * HEADS + h])) /
                      (denom[d * HEADS + h] + 1e-16f);
        int cl = c0 > h * C ? c0 : h * C;
        int ch = (c0 + 32) < (h + 1) * C ? (c0 + 32) : (h + 1) * C;
        for (int c = cl; c < ch; ++c)
            atomicAdd(&out[(size_t)d * FDIM + c],
                      alpha * H[(size_t)s * FDIM + c]);
    }
}

__global__ void bias_relu_kernel(float* __restrict__ X,
                                 const float* __restrict__ b) {
    int t = blockIdx.x * blockDim.x + threadIdx.x;
    if (t >= NN * FDIM) return;
    float v = X[t] + b[t & (FDIM - 1)];
    X[t] = v > 0.f ? v : 0.f;
}

// ---- readout:  out = (colsum(h0) + colsum(h5)) @ lin_W ----
__global__ void zero_cs_kernel(float* __restrict__ cs) {
    if (threadIdx.x < FDIM) cs[threadIdx.x] = 0.f;
}
__global__ void colsum_kernel(const float* __restrict__ X,
                              float* __restrict__ cs, int rows_per_block) {
    int c  = threadIdx.x;                        // 0..127, coalesced
    int n0 = blockIdx.x * rows_per_block;
    int n1 = n0 + rows_per_block; if (n1 > NN) n1 = NN;
    float s = 0.f;
    for (int n = n0; n < n1; ++n) s += X[(size_t)n * FDIM + c];
    atomicAdd(&cs[c], s);
}
__global__ void readout_kernel(const float* __restrict__ cs,
                               const float* __restrict__ linW,
                               float* __restrict__ out) {
    int j = threadIdx.x;                         // 0..63
    if (j >= 64) return;
    float s = 0.f;
#pragma unroll 4
    for (int c = 0; c < FDIM; ++c) s += cs[c] * linW[c * 64 + j];
    out[j] = s;
}

// ============================================================
extern "C" void kernel_launch(void* const* d_in, const int* in_sizes, int n_in,
                              void* d_out, int out_size, void* d_ws, size_t ws_size,
                              hipStream_t stream) {
    const float* h0 = (const float*)d_in[0];
    const int*   ei = (const int*)d_in[1];
    // params flattened in dict insertion order: (W, as, ad, b) x5, then lin_W
    const float* linW = (const float*)d_in[22];

    // -------- workspace carve-up (~83 MB total) --------
    char* ws = (char*)d_ws; size_t off = 0;
    auto carve = [&](size_t bytes) -> void* {
        void* p = ws + off; off = (off + bytes + 255) & ~(size_t)255; return p;
    };
    float*     bufA  = (float*)carve((size_t)NN * FDIM * 4);
    float*     bufB  = (float*)carve((size_t)NN * FDIM * 4);
    float*     gout  = (float*)carve((size_t)NN * FDIM * 4);
    float*     a_s   = (float*)carve((size_t)NN * 8 * 4);
    float*     a_d   = (float*)carve((size_t)NN * 8 * 4);
    unsigned*  mbuf  = (unsigned*)carve((size_t)NN * 8 * 4);
    float*     denom = (float*)carve((size_t)NN * 8 * 4);
    _Float16*  Wp    = (_Float16*)carve(8 * 4 * 32 * 16 * 2);
    float*     cs    = (float*)carve(FDIM * 4);

    const int TB = 256;
    const int gNF   = (NN * FDIM + TB - 1) / TB;   // node*feature grid
    const int gE    = (ETOT + TB - 1) / TB;        // edge grid
    const int gE4   = (ETOT * 4 + TB - 1) / TB;    // edge*4 grid
    const int gGemm = NN / 16;                     // 3125, exact

    const float* in  = h0;
    float*       out = bufA;
    for (int l = 0; l < 5; ++l) {
        const float* W  = (const float*)d_in[2 + 4 * l];
        const float* as = (const float*)d_in[3 + 4 * l];
        const float* ad = (const float*)d_in[4 + 4 * l];
        const float* b  = (const float*)d_in[5 + 4 * l];

        pack_w_kernel<<<4, TB, 0, stream>>>(W, Wp);
        gemm128_kernel<<<gGemm, TB, 0, stream>>>(in, Wp, gout);

        if (l < 4) {   // 8 heads x 16 channels
            att_scores_kernel<8, 16><<<(NN * 8 + TB - 1) / TB, TB, 0, stream>>>(
                gout, as, ad, a_s, a_d);
            init_layer_kernel<8><<<gNF, TB, 0, stream>>>(mbuf, denom, out);
            edge_max_kernel<8><<<gE, TB, 0, stream>>>(ei, a_s, a_d, mbuf);
            edge_denom_kernel<8><<<gE, TB, 0, stream>>>(ei, a_s, a_d, mbuf, denom);
            edge_scatter_kernel<8, 16><<<gE4, TB, 0, stream>>>(
                ei, a_s, a_d, mbuf, denom, gout, out);
        } else {       // 1 head x 128 channels (mean over 1 head == identity)
            att_scores_kernel<1, 128><<<(NN + TB - 1) / TB, TB, 0, stream>>>(
                gout, as, ad, a_s, a_d);
            init_layer_kernel<1><<<gNF, TB, 0, stream>>>(mbuf, denom, out);
            edge_max_kernel<1><<<gE, TB, 0, stream>>>(ei, a_s, a_d, mbuf);
            edge_denom_kernel<1><<<gE, TB, 0, stream>>>(ei, a_s, a_d, mbuf, denom);
            edge_scatter_kernel<1, 128><<<gE4, TB, 0, stream>>>(
                ei, a_s, a_d, mbuf, denom, gout, out);
        }
        bias_relu_kernel<<<gNF, TB, 0, stream>>>(out, b);

        in  = out;
        out = (out == bufA) ? bufB : bufA;
    }
    const float* hfinal = in;   // bufA after 5 layers

    zero_cs_kernel<<<1, FDIM, 0, stream>>>(cs);
    colsum_kernel<<<(NN + 127) / 128, FDIM, 0, stream>>>(h0, cs, 128);
    colsum_kernel<<<(NN + 127) / 128, FDIM, 0, stream>>>(hfinal, cs, 128);
    readout_kernel<<<1, 64, 0, stream>>>(cs, linW, (float*)d_out);
}
